// GNNModel_8443905704148
// MI455X (gfx1250) — matrix operand order repaired and checked by reading, not compiled, not run on hardware
//
#include <hip/hip_runtime.h>
#include <hip/hip_bf16.h>
#include <math.h>

typedef __attribute__((ext_vector_type(16))) _Float16 v16h;
typedef __attribute__((ext_vector_type(8)))  _Float16 v8h;
typedef __attribute__((ext_vector_type(8)))  float    v8f;

#define N_NODES 50000      // 3125 * 16, exact
#define N_EDGES 800000     // 12500 * 64, exact
#define ORIG_F  64
#define EDGE_F  64
#define NODE_F  96
#define H_F     128
#define CONV_K  (2*NODE_F + EDGE_F)   // 256
#define EPB     64                    // edges per conv block
#define ZSTRIDE (CONV_K + 8)          // 264 halfs = 528 B: +4 banks/row -> conflict-free

__device__ __forceinline__ v8f wmma_f16(v16h a, v16h b, v8f c) {
    return __builtin_amdgcn_wmma_f32_16x16x32_f16(
        false, a, false, b, (short)0, c, false, false);
}

// A fragment (ISA 7.12.2): elems 0..7 = K[kb+half*8 ..], elems 8..15 = K[kb+16+half*8 ..]
__device__ __forceinline__ v16h combine_a(v8h lo, v8h hi) {
    return __builtin_shufflevector(lo, hi, 0,1,2,3,4,5,6,7,8,9,10,11,12,13,14,15);
}

// per-lane async 16B copy global -> LDS (GV addressing), tracked by ASYNCcnt
__device__ __forceinline__ void async_copy_b128(void* lds_dst, const void* gsrc) {
    unsigned lds = (unsigned)(uintptr_t)lds_dst;   // low 32 bits = LDS offset
    unsigned long long ga = (unsigned long long)(uintptr_t)gsrc;
    asm volatile("global_load_async_to_lds_b128 %0, %1, off"
                 :: "v"(lds), "v"(ga) : "memory");
}
__device__ __forceinline__ void wait_asynccnt0() {
    asm volatile("s_wait_asynccnt 0x0" ::: "memory");
}

// ---------------------------------------------------------------------------
// Pack W[k][n] (f32, KxN) into B-fragment layout:
//   Wf[(((k>>5)*2 + ((k>>3)&1))*N + n)*16 + (k&7) + ((k>>4)&1)*8] = (f16) W[k*N + n]
// ---------------------------------------------------------------------------
__global__ void pack_w16(const float* __restrict__ W, _Float16* __restrict__ Wf,
                         int K, int N) {
    int i = blockIdx.x * blockDim.x + threadIdx.x;
    if (i < K * N) {
        int k = i / N, n = i - k * N;
        Wf[(((k >> 5) * 2 + ((k >> 3) & 1)) * N + n) * 16 + (k & 7) + ((k >> 4) & 1) * 8] = (_Float16)W[i];
    }
}

// ---------------------------------------------------------------------------
// h16 = f16( x @ emb_W + emb_b )   [50000,64] x [64,96]; 6 waves/blk, grid 3125
// ---------------------------------------------------------------------------
__global__ void embed_gemm(const float* __restrict__ x, const _Float16* __restrict__ Wf,
                           const float* __restrict__ bias, _Float16* __restrict__ h16) {
    const int lane = threadIdx.x & 31, wv = threadIdx.x >> 5;   // wv 0..5
    const int half = lane >> 4, l15 = lane & 15;
    const int tm = blockIdx.x;
    const int row = tm * 16 + l15;
    const int col = wv * 16 + l15;

    v8f c = {};
    #pragma unroll
    for (int kb = 0; kb < ORIG_F; kb += 32) {
        v16h b = *(const v16h*)(Wf + (((kb >> 4) + half) * NODE_F + col) * 16);
        v8f x0 = __builtin_nontemporal_load((const v8f*)(x + row * ORIG_F + kb + half * 8));
        v8f x1 = __builtin_nontemporal_load((const v8f*)(x + row * ORIG_F + kb + 16 + half * 8));
        v16h a;
        #pragma unroll
        for (int t = 0; t < 8; ++t) { a[t] = (_Float16)x0[t]; a[8+t] = (_Float16)x1[t]; }
        c = wmma_f16(a, b, c);
    }
    const float bv = bias[col];
    #pragma unroll
    for (int v = 0; v < 8; ++v) {
        int r = tm * 16 + v + half * 8;
        h16[r * NODE_F + col] = (_Float16)(c[v] + bv);
    }
}

// ---------------------------------------------------------------------------
// Edge conv: m = relu([h16[dst] | h16[src] | ea] @ W + b); atomic scatter to dst.
// Block = 64 edges, 192 threads (6 waves). LDS tile 64 x 264 f16 (~33 KB, padded).
// h-row chunks staged with GLOBAL_LOAD_ASYNC_TO_LDS_B128; ea chunks cvt'd in VALU.
// ---------------------------------------------------------------------------
__global__ void conv_gemm_scatter(const _Float16* __restrict__ h16,
                                  const int* __restrict__ esrc, const int* __restrict__ edst,
                                  const float* __restrict__ ea,
                                  const _Float16* __restrict__ Wf, const float* __restrict__ bias,
                                  float* __restrict__ accum) {
    __shared__ __align__(16) _Float16 zt[EPB * ZSTRIDE];
    __shared__ int sdst[EPB], ssrc[EPB];
    const int tid = threadIdx.x;
    const int e0  = blockIdx.x * EPB;

    if (tid < EPB) {
        sdst[tid] = edst[e0 + tid];
        ssrc[tid] = esrc[e0 + tid];
    }
    __syncthreads();

    // 32 16B-chunks per row: 12 from h16[dst] | 12 from h16[src] | 8 from ea
    for (int cidx = tid; cidx < EPB * 32; cidx += 192) {
        int r = cidx >> 5, cc = cidx & 31;
        _Float16* dst = zt + r * ZSTRIDE + cc * 8;
        if (cc < 24) {
            int node = (cc < 12) ? sdst[r] : ssrc[r];
            int koff = (cc < 12) ? cc * 8 : (cc - 12) * 8;
            async_copy_b128(dst, h16 + node * NODE_F + koff);
        } else {
            v8f ev = __builtin_nontemporal_load(
                (const v8f*)(ea + (e0 + r) * EDGE_F + (cc - 24) * 8));
            v8h val;
            #pragma unroll
            for (int t = 0; t < 8; ++t) val[t] = (_Float16)ev[t];
            *(v8h*)dst = val;
        }
    }
    wait_asynccnt0();
    __syncthreads();

    const int lane = tid & 31, wv = tid >> 5;   // wv 0..5 -> col tile
    const int half = lane >> 4, l15 = lane & 15;
    const int col  = wv * 16 + l15;

    v8f c0 = {}, c1 = {}, c2 = {}, c3 = {};
    for (int kb = 0; kb < CONV_K; kb += 32) {
        v16h b = *(const v16h*)(Wf + (((kb >> 4) + half) * NODE_F + col) * 16);
        const _Float16* z0 = zt + (l15)      * ZSTRIDE + kb + half * 8;
        const _Float16* z1 = zt + (16 + l15) * ZSTRIDE + kb + half * 8;
        const _Float16* z2 = zt + (32 + l15) * ZSTRIDE + kb + half * 8;
        const _Float16* z3 = zt + (48 + l15) * ZSTRIDE + kb + half * 8;
        c0 = wmma_f16(combine_a(*(const v8h*)z0, *(const v8h*)(z0 + 16)), b, c0);
        c1 = wmma_f16(combine_a(*(const v8h*)z1, *(const v8h*)(z1 + 16)), b, c1);
        c2 = wmma_f16(combine_a(*(const v8h*)z2, *(const v8h*)(z2 + 16)), b, c2);
        c3 = wmma_f16(combine_a(*(const v8h*)z3, *(const v8h*)(z3 + 16)), b, c3);
    }

    const float bv = bias[col];
    #pragma unroll
    for (int mt = 0; mt < 4; ++mt) {
        v8f cc = (mt == 0) ? c0 : (mt == 1) ? c1 : (mt == 2) ? c2 : c3;
        #pragma unroll
        for (int v = 0; v < 8; ++v) {
            int r = mt * 16 + v + half * 8;
            float m = cc[v] + bv;
            m = m > 0.f ? m : 0.f;
            atomicAdd(&accum[sdst[r] * NODE_F + col], m);
        }
    }
}

// ---------------------------------------------------------------------------
// hp = softplus( h16 @ post_W + post_b )  [50000,96] x [96,128]; 8 waves/blk
// ---------------------------------------------------------------------------
__global__ void post_gemm(const _Float16* __restrict__ h16, const _Float16* __restrict__ Wf,
                          const float* __restrict__ bias, float* __restrict__ hp) {
    const int lane = threadIdx.x & 31, wv = threadIdx.x >> 5;  // wv 0..7
    const int half = lane >> 4, l15 = lane & 15;
    const int tm = blockIdx.x;
    const int row = tm * 16 + l15;
    const int col = wv * 16 + l15;

    v8f c = {};
    #pragma unroll
    for (int kb = 0; kb < NODE_F; kb += 32) {
        v16h b = *(const v16h*)(Wf + (((kb >> 4) + half) * H_F + col) * 16);
        const _Float16* z = h16 + row * NODE_F + kb + half * 8;
        c = wmma_f16(combine_a(*(const v8h*)z, *(const v8h*)(z + 16)), b, c);
    }
    const float bv = bias[col];
    #pragma unroll
    for (int v = 0; v < 8; ++v) {
        int r = tm * 16 + v + half * 8;
        float z = c[v] + bv;
        hp[r * H_F + col] = (z > 20.f) ? z : log1pf(expf(z));
    }
}

// ---------------------------------------------------------------------------
__global__ void zero_kernel(float* __restrict__ p, int n) {
    int i = blockIdx.x * blockDim.x + threadIdx.x;
    if (i < n) p[i] = 0.f;
}

__global__ void deg_kernel(const int* __restrict__ edst, float* __restrict__ deg) {
    int e = blockIdx.x * blockDim.x + threadIdx.x;
    if (e < N_EDGES) atomicAdd(&deg[edst[e]], 1.0f);
}

__global__ void mean_kernel(const float* __restrict__ accum, const float* __restrict__ deg,
                            _Float16* __restrict__ h16) {
    int i = blockIdx.x * blockDim.x + threadIdx.x;
    if (i < N_NODES * NODE_F) {
        float d = fmaxf(deg[i / NODE_F], 1.0f);
        h16[i] = (_Float16)(accum[i] / d);
    }
}

__global__ void pool_sum(const float* __restrict__ hp, const int* __restrict__ batch,
                         float* __restrict__ gsum, float* __restrict__ gcnt) {
    int i = blockIdx.x * blockDim.x + threadIdx.x;
    if (i < N_NODES * H_F) {
        int node = i >> 7, k = i & 127;
        int g = batch[node];
        atomicAdd(&gsum[g * H_F + k], hp[i]);
        if (k == 0) atomicAdd(&gcnt[g], 1.0f);
    }
}

__global__ void head_kernel(const float* __restrict__ gsum, const float* __restrict__ gcnt,
                            const float* __restrict__ hidW, const float* __restrict__ hidB,
                            const float* __restrict__ f1W, const float* __restrict__ f1B,
                            const float* __restrict__ f2W, const float* __restrict__ f2B,
                            const float* __restrict__ f3W, const float* __restrict__ f3B,
                            float* __restrict__ out) {
    __shared__ float gs[H_F], t1[H_F], t2[8], t3[4];
    const int g = blockIdx.x, t = threadIdx.x;
    float cnt = fmaxf(gcnt[g], 1.0f);
    gs[t] = gsum[g * H_F + t] / cnt;
    __syncthreads();
    float acc = hidB[t];
    for (int i = 0; i < H_F; ++i) acc += gs[i] * hidW[i * H_F + t];
    t1[t] = fmaxf(acc, 0.f);
    __syncthreads();
    if (t < 8) {
        float a = f1B[t];
        for (int i = 0; i < H_F; ++i) a += t1[i] * f1W[i * 8 + t];
        t2[t] = fmaxf(a, 0.f);
    }
    __syncthreads();
    if (t < 4) {
        float a = f2B[t];
        for (int i = 0; i < 8; ++i) a += t2[i] * f2W[i * 4 + t];
        t3[t] = fmaxf(a, 0.f);
    }
    __syncthreads();
    if (t == 0) {
        float a = f3B[0];
        for (int i = 0; i < 4; ++i) a += t3[i] * f3W[i];
        out[g] = a;
    }
}

// ---------------------------------------------------------------------------
extern "C" void kernel_launch(void* const* d_in, const int* in_sizes, int n_in,
                              void* d_out, int out_size, void* d_ws, size_t ws_size,
                              hipStream_t stream) {
    (void)in_sizes; (void)n_in; (void)out_size; (void)ws_size;

    const float* x     = (const float*)d_in[0];
    const int*   ei    = (const int*)d_in[1];
    const float* ea    = (const float*)d_in[2];
    const int*   batch = (const int*)d_in[3];
    const float* embW  = (const float*)d_in[4];
    const float* embB  = (const float*)d_in[5];
    const float* convW = (const float*)d_in[6];
    const float* convB = (const float*)d_in[7];
    const float* postW = (const float*)d_in[8];
    const float* postB = (const float*)d_in[9];
    const float* hidW  = (const float*)d_in[10];
    const float* hidB  = (const float*)d_in[11];
    const float* f1W   = (const float*)d_in[12];
    const float* f1B   = (const float*)d_in[13];
    const float* f2W   = (const float*)d_in[14];
    const float* f2B   = (const float*)d_in[15];
    const float* f3W   = (const float*)d_in[16];
    const float* f3B   = (const float*)d_in[17];
    float* out = (float*)d_out;

    const int* esrc = ei;
    const int* edst = ei + N_EDGES;

    char* p = (char*)d_ws;
    auto carve = [&](size_t bytes) { char* r = p; p += (bytes + 255) & ~(size_t)255; return r; };
    float*    accum  = (float*)   carve((size_t)N_NODES * NODE_F * 4);
    float*    deg    = (float*)   carve((size_t)N_NODES * 4);
    float*    hp     = (float*)   carve((size_t)N_NODES * H_F * 4);
    float*    gsum   = (float*)   carve((size_t)(64 * H_F + 64) * 4);
    float*    gcnt   = gsum + 64 * H_F;
    _Float16* h16    = (_Float16*)carve((size_t)N_NODES * NODE_F * 2);
    _Float16* embWf  = (_Float16*)carve((size_t)ORIG_F * NODE_F * 2);
    _Float16* convWf = (_Float16*)carve((size_t)2 * CONV_K * NODE_F * 2);
    _Float16* postWf = (_Float16*)carve((size_t)NODE_F * H_F * 2);

    // pack weights into B-fragment f16 layout
    pack_w16<<<(ORIG_F * NODE_F + 255) / 256, 256, 0, stream>>>(embW, embWf, ORIG_F, NODE_F);
    pack_w16<<<(2 * CONV_K * NODE_F + 255) / 256, 256, 0, stream>>>(convW, convWf, 2 * CONV_K, NODE_F);
    pack_w16<<<(NODE_F * H_F + 255) / 256, 256, 0, stream>>>(postW, postWf, NODE_F, H_F);

    // degrees (dst is layer-invariant)
    zero_kernel<<<(N_NODES + 255) / 256, 256, 0, stream>>>(deg, N_NODES);
    deg_kernel<<<(N_EDGES + 255) / 256, 256, 0, stream>>>(edst, deg);

    // node embedding -> h16
    embed_gemm<<<N_NODES / 16, 192, 0, stream>>>(x, embWf, embB, h16);

    // two conv layers
    for (int layer = 0; layer < 2; ++layer) {
        zero_kernel<<<(N_NODES * NODE_F + 255) / 256, 256, 0, stream>>>(accum, N_NODES * NODE_F);
        conv_gemm_scatter<<<N_EDGES / EPB, 192, 0, stream>>>(
            h16, esrc, edst, ea,
            convWf + (size_t)layer * CONV_K * NODE_F, convB + layer * NODE_F, accum);
        mean_kernel<<<(N_NODES * NODE_F + 255) / 256, 256, 0, stream>>>(accum, deg, h16);
    }

    // post linear + softplus -> hp (f32)
    post_gemm<<<N_NODES / 16, 256, 0, stream>>>(h16, postWf, postB, hp);

    // global mean pool + head MLP
    zero_kernel<<<(64 * H_F + 64 + 255) / 256, 256, 0, stream>>>(gsum, 64 * H_F + 64);
    pool_sum<<<(N_NODES * H_F + 255) / 256, 256, 0, stream>>>(hp, batch, gsum, gcnt);
    head_kernel<<<64, 128, 0, stream>>>(gsum, gcnt, hidW, hidB, f1W, f1B,
                                        f2W, f2B, f3W, f3B, out);
}
